// RelativeMHAttention_58935541236341
// MI455X (gfx1250) — matrix operand order repaired and checked by reading, not compiled, not run on hardware
//
#include <hip/hip_runtime.h>

// ---- problem constants (fixed by the reference) ----
#define BATCH   2
#define S_LEN   2048
#define DMODEL  1024
#define NHEAD   16
#define DHEAD   64
#define MAXREL  2
#define NBUCKET 5
#define MROWS   (BATCH * S_LEN)   // 4096

typedef __attribute__((ext_vector_type(16))) __bf16 v16bf;
typedef __attribute__((ext_vector_type(8)))  float  v8f;

union AF { v16bf v; unsigned int u[8]; };

// A-matrix (16x32 bf16) dword->K mapping: lanes0-15 K={0..7,16..23}, lanes16-31 K={8..15,24..31}
__device__ __forceinline__ int kbA(int j, int hlf) {
    return 2 * j + ((j >= 4) ? 8 : 0) + 8 * hlf;
}
// B-matrix (32x16 bf16) dword->K mapping: lanes0-15 K={0..7,16..23}, lanes16-31 K={8..15,24..31}
__device__ __forceinline__ int kbB(int j, int hlf) {
    return 2 * j + ((j >= 4) ? 8 : 0) + 8 * hlf;
}

// ---- gfx1250 async global->LDS copy (ASYNCcnt path, ISA 08_async_tensor §4) ----
// LDS flat addresses have the byte offset in the low 32 bits (aperture in [63:32]),
// so truncating the generic pointer yields the LDS address VGPR operand.
__device__ __forceinline__ void async_ld_b128(void* lds, const void* gptr) {
    unsigned l = (unsigned)(unsigned long long)lds;
    unsigned long long g = (unsigned long long)gptr;
    asm volatile("global_load_async_to_lds_b128 %0, %1, off"
                 :: "v"(l), "v"(g) : "memory");
}
__device__ __forceinline__ void wait_async() {
    asm volatile("s_wait_asynccnt 0x0" ::: "memory");
}

// ---------------------------------------------------------------------------
// fp32 -> bf16 conversion (vectorized: 4 elements / thread)
// ---------------------------------------------------------------------------
__global__ void cvt_bf16_kernel(const float4* __restrict__ src,
                                uint2* __restrict__ dst, int n4) {
    int i = blockIdx.x * blockDim.x + threadIdx.x;
    if (i < n4) {
        float4 f = src[i];
        union { __bf16 h[4]; uint2 u; } b;
        b.h[0] = (__bf16)f.x; b.h[1] = (__bf16)f.y;
        b.h[2] = (__bf16)f.z; b.h[3] = (__bf16)f.w;
        dst[i] = b.u;
    }
}

// ---------------------------------------------------------------------------
// QKV projection GEMM: X[4096x1024] @ W[1024x1024] + b -> [B,H,S,D] bf16
// block = 128 threads (4 waves, 2x2), tile 64x64, K-step 32, double-buffered
// LDS with async A staging. blockIdx.z selects Q / K / V.
// ---------------------------------------------------------------------------
__global__ __launch_bounds__(128)
void gemm_qkv_kernel(const __bf16* __restrict__ X,
                     const __bf16* __restrict__ Wq, const __bf16* __restrict__ Wk,
                     const __bf16* __restrict__ Wv,
                     const float* __restrict__ bq, const float* __restrict__ bk,
                     const float* __restrict__ bv,
                     __bf16* __restrict__ Qo, __bf16* __restrict__ Ko,
                     __bf16* __restrict__ Vo) {
    const __bf16* W; const float* bias; __bf16* Out;
    if (blockIdx.z == 0)      { W = Wq; bias = bq; Out = Qo; }
    else if (blockIdx.z == 1) { W = Wk; bias = bk; Out = Ko; }
    else                      { W = Wv; bias = bv; Out = Vo; }

    __shared__ __bf16 ldsA[2][64 * 32];
    __shared__ __bf16 ldsBt[2][64 * 32];   // transposed: [n][k]

    const int tid  = threadIdx.x;
    const int lane = tid & 31, wave = tid >> 5;
    const int wm = wave >> 1, wn = wave & 1;
    const int hlf = lane >> 4, l16 = lane & 15;
    const int m0 = blockIdx.x * 64, n0 = blockIdx.y * 64;

    v8f acc[2][2] = {};

    auto stage = [&](int buf, int k0) {
        { // A tile 64x32 row-major: async global->LDS
            int row = tid >> 1, h2 = tid & 1;
            const __bf16* gA = X + (size_t)(m0 + row) * DMODEL + k0 + h2 * 16;
            __bf16* lA = &ldsA[buf][row * 32 + h2 * 16];
            async_ld_b128(lA, gA);
            async_ld_b128(lA + 8, gA + 8);
        }
        { // B tile 32x64 transposed into [n][k] (manual scatter)
            int kr = tid >> 2, qt = tid & 3;
            union { uint4 q[2]; __bf16 h[16]; } bufr;
            const uint4* src = (const uint4*)(W + (size_t)(k0 + kr) * DMODEL + n0 + qt * 16);
            bufr.q[0] = src[0]; bufr.q[1] = src[1];
#pragma unroll
            for (int e = 0; e < 16; ++e)
                ldsBt[buf][(qt * 16 + e) * 32 + kr] = bufr.h[e];
        }
    };

    stage(0, 0);
    wait_async();
    __syncthreads();

    for (int k0 = 0; k0 < DMODEL; k0 += 32) {
        const int cur = (k0 >> 5) & 1;
        if (k0 + 32 < DMODEL) stage(1 - cur, k0 + 32);   // overlap with compute

        AF a[2], b[2];
#pragma unroll
        for (int sm = 0; sm < 2; ++sm) {
            int mr = wm * 32 + sm * 16 + l16;
#pragma unroll
            for (int j = 0; j < 8; ++j)
                a[sm].u[j] = *(const unsigned int*)&ldsA[cur][mr * 32 + kbA(j, hlf)];
        }
#pragma unroll
        for (int sn = 0; sn < 2; ++sn) {
            int nc = wn * 32 + sn * 16 + l16;
#pragma unroll
            for (int j = 0; j < 8; ++j)
                b[sn].u[j] = *(const unsigned int*)&ldsBt[cur][nc * 32 + kbB(j, hlf)];
        }
#pragma unroll
        for (int sm = 0; sm < 2; ++sm)
#pragma unroll
            for (int sn = 0; sn < 2; ++sn)
                acc[sm][sn] = __builtin_amdgcn_wmma_f32_16x16x32_bf16(
                    false, a[sm].v, false, b[sn].v, (short)0, acc[sm][sn], false, false);

        wait_async();
        __syncthreads();
    }

    // scatter to [B,H,S,D] bf16 with bias
#pragma unroll
    for (int sm = 0; sm < 2; ++sm)
#pragma unroll
        for (int sn = 0; sn < 2; ++sn)
#pragma unroll
            for (int r = 0; r < 8; ++r) {
                int m = m0 + wm * 32 + sm * 16 + r + 8 * hlf;
                int n = n0 + wn * 32 + sn * 16 + l16;
                int bI = m >> 11, sI = m & (S_LEN - 1);
                int hI = n >> 6,  dI = n & (DHEAD - 1);
                float v = acc[sm][sn][r] + bias[n];
                Out[(((size_t)(bI * NHEAD + hI) * S_LEN) + sI) * DHEAD + dI] = (__bf16)v;
            }
}

// ---------------------------------------------------------------------------
// Flash-style attention with relative-position bias + bucket value term.
// grid = (S/64, B*H), block = 128 (4 waves; wave w owns queries [q0+16w, +16))
// Double-buffered K/V tiles; K staged with async global->LDS.
// ---------------------------------------------------------------------------
__global__ __launch_bounds__(128)
void attn_kernel(const __bf16* __restrict__ Q, const __bf16* __restrict__ K,
                 const __bf16* __restrict__ V,
                 const float* __restrict__ rel_k, const float* __restrict__ rel_v,
                 __bf16* __restrict__ Ctx) {
    const int tid = threadIdx.x, lane = tid & 31, wave = tid >> 5;
    const int hlf = lane >> 4, l16 = lane & 15;
    const int bh = blockIdx.y;
    const int bI = bh >> 4, hI = bh & (NHEAD - 1);
    const int q0 = blockIdx.x * 64;

    const size_t base = (size_t)bh * S_LEN * DHEAD;
    const __bf16* Qp = Q + base;
    const __bf16* Kp = K + base;
    const __bf16* Vp = V + base;

    __shared__ __bf16 ldsK[2][64 * 64];       // [key][d]
    __shared__ __bf16 ldsVt[2][64 * 64];      // [d][key]
    __shared__ __bf16 ldsP[4 * 16 * 64];      // per-wave P tile [q][k]
    __shared__ float  ldsQR[64 * NBUCKET];    // q . rel_k[j]
    __shared__ float  ldsRV[NBUCKET * DHEAD];

    for (int e = tid; e < NBUCKET * DHEAD; e += 128) ldsRV[e] = rel_v[e];
    for (int e = tid; e < 64 * NBUCKET; e += 128) {
        int qr = e / NBUCKET, j = e % NBUCKET;
        const __bf16* qrow = Qp + (size_t)(q0 + qr) * DHEAD;
        float s = 0.f;
        for (int d = 0; d < DHEAD; ++d) s += (float)qrow[d] * rel_k[j * DHEAD + d];
        ldsQR[e] = s;
    }

    // Q fragments for this wave (A-matrix role), kept in registers
    AF qa[2];
    {
        const __bf16* qrow = Qp + (size_t)(q0 + wave * 16 + l16) * DHEAD;
#pragma unroll
        for (int dc = 0; dc < 2; ++dc)
#pragma unroll
            for (int j = 0; j < 8; ++j)
                qa[dc].u[j] = *(const unsigned int*)&qrow[dc * 32 + kbA(j, hlf)];
    }

    v8f oacc[4] = {};
    float mI[8], lI[8], bk5[8][NBUCKET];
#pragma unroll
    for (int r = 0; r < 8; ++r) {
        mI[r] = -1e30f; lI[r] = 0.f;
#pragma unroll
        for (int j = 0; j < NBUCKET; ++j) bk5[r][j] = 0.f;
    }
    const int qg_base = q0 + wave * 16;

    auto stage = [&](int buf, int kt) {
        int key = tid >> 1, dh = tid & 1;
        { // K tile [key][d]: async global->LDS, 64B per thread
            const __bf16* gK = Kp + (size_t)(kt * 64 + key) * DHEAD + dh * 32;
            __bf16* lK = &ldsK[buf][key * 64 + dh * 32];
#pragma unroll
            for (int c = 0; c < 4; ++c)
                async_ld_b128(lK + c * 8, gK + c * 8);
        }
        { // V tile transposed [d][key] (manual scatter)
            union { uint4 q[4]; __bf16 h[32]; } bufr;
            const uint4* src = (const uint4*)(Vp + (size_t)(kt * 64 + key) * DHEAD + dh * 32);
            bufr.q[0] = src[0]; bufr.q[1] = src[1]; bufr.q[2] = src[2]; bufr.q[3] = src[3];
#pragma unroll
            for (int e = 0; e < 32; ++e)
                ldsVt[buf][(dh * 32 + e) * 64 + key] = bufr.h[e];
        }
    };

    stage(0, 0);
    wait_async();
    __syncthreads();

    for (int kt = 0; kt < S_LEN / 64; ++kt) {
        const int cur = kt & 1;
        if (kt + 1 < S_LEN / 64) stage(1 - cur, kt + 1);   // overlap with compute

        // ---- S = Q Kt^T (4 key subtiles, 2 d-chunks each) ----
        v8f sacc[4];
#pragma unroll
        for (int sn = 0; sn < 4; ++sn) {
            v8f z = {};
#pragma unroll
            for (int dc = 0; dc < 2; ++dc) {
                AF bfr;
#pragma unroll
                for (int j = 0; j < 8; ++j)
                    bfr.u[j] = *(const unsigned int*)
                        &ldsK[cur][(sn * 16 + l16) * 64 + dc * 32 + kbA(j, hlf)];
                z = __builtin_amdgcn_wmma_f32_16x16x32_bf16(
                    false, qa[dc].v, false, bfr.v, (short)0, z, false, false);
            }
            sacc[sn] = z;
        }

        // ---- relative bias, scale, running max ----
        float mt[8];
#pragma unroll
        for (int r = 0; r < 8; ++r) mt[r] = -1e30f;
#pragma unroll
        for (int sn = 0; sn < 4; ++sn) {
            int kg = kt * 64 + sn * 16 + l16;
#pragma unroll
            for (int r = 0; r < 8; ++r) {
                int qrow = r + 8 * hlf;
                int delta = kg - (qg_base + qrow);
                int bkt = delta < -MAXREL ? 0 : (delta > MAXREL ? 2 * MAXREL : delta + MAXREL);
                float v = (sacc[sn][r] + ldsQR[(wave * 16 + qrow) * NBUCKET + bkt]) * 0.125f;
                sacc[sn][r] = v;
                mt[r] = fmaxf(mt[r], v);
            }
        }
#pragma unroll
        for (int r = 0; r < 8; ++r) {
            float v = mt[r];
            v = fmaxf(v, __shfl_xor(v, 1, 32));
            v = fmaxf(v, __shfl_xor(v, 2, 32));
            v = fmaxf(v, __shfl_xor(v, 4, 32));
            v = fmaxf(v, __shfl_xor(v, 8, 32));
            mt[r] = v;
        }
        float alpha[8], rs[8];
#pragma unroll
        for (int r = 0; r < 8; ++r) {
            float mn = fmaxf(mI[r], mt[r]);
            alpha[r] = __expf(mI[r] - mn);
            mI[r] = mn;
            rs[r] = 0.f;
#pragma unroll
            for (int j = 0; j < NBUCKET; ++j) bk5[r][j] *= alpha[r];
        }
        // ---- P = exp(S - m); row sums, bucket sums; spill P to LDS as bf16 ----
#pragma unroll
        for (int sn = 0; sn < 4; ++sn) {
            int kg = kt * 64 + sn * 16 + l16;
#pragma unroll
            for (int r = 0; r < 8; ++r) {
                int qrow = r + 8 * hlf;
                int delta = kg - (qg_base + qrow);
                int bkt = delta < -MAXREL ? 0 : (delta > MAXREL ? 2 * MAXREL : delta + MAXREL);
                float p = __expf(sacc[sn][r] - mI[r]);
                rs[r] += p;
                bk5[r][bkt] += p;
                ldsP[wave * 1024 + qrow * 64 + sn * 16 + l16] = (__bf16)p;
            }
        }
#pragma unroll
        for (int r = 0; r < 8; ++r) {
            float v = rs[r];
            v += __shfl_xor(v, 1, 32);
            v += __shfl_xor(v, 2, 32);
            v += __shfl_xor(v, 4, 32);
            v += __shfl_xor(v, 8, 32);
            lI[r] = lI[r] * alpha[r] + v;
        }
#pragma unroll
        for (int dt = 0; dt < 4; ++dt)
#pragma unroll
            for (int r = 0; r < 8; ++r)
                oacc[dt][r] *= alpha[r];

        __syncthreads();  // P visible cross-lane

        // ---- O += P * V ----
#pragma unroll
        for (int kc = 0; kc < 2; ++kc) {
            AF pa;
#pragma unroll
            for (int j = 0; j < 8; ++j)
                pa.u[j] = *(const unsigned int*)
                    &ldsP[wave * 1024 + l16 * 64 + kc * 32 + kbA(j, hlf)];
#pragma unroll
            for (int dt = 0; dt < 4; ++dt) {
                AF bvr;
#pragma unroll
                for (int j = 0; j < 8; ++j)
                    bvr.u[j] = *(const unsigned int*)
                        &ldsVt[cur][(dt * 16 + l16) * 64 + kc * 32 + kbA(j, hlf)];
                oacc[dt] = __builtin_amdgcn_wmma_f32_16x16x32_bf16(
                    false, pa.v, false, bvr.v, (short)0, oacc[dt], false, false);
            }
        }

        wait_async();      // next tile's async K copies landed
        __syncthreads();   // everyone done with ldsP / buffers before next iter
    }

    // bucket sums: reduce across the 16 lanes holding a row's columns
#pragma unroll
    for (int r = 0; r < 8; ++r)
#pragma unroll
        for (int j = 0; j < NBUCKET; ++j) {
            float v = bk5[r][j];
            v += __shfl_xor(v, 1, 32);
            v += __shfl_xor(v, 2, 32);
            v += __shfl_xor(v, 4, 32);
            v += __shfl_xor(v, 8, 32);
            bk5[r][j] = v;
        }

    // normalize, add bucket * rel_v, store context (bf16, [B,S,Dm] layout)
#pragma unroll
    for (int r = 0; r < 8; ++r) {
        int qrow = r + 8 * hlf;
        int qg = qg_base + qrow;
        float inv = 1.f / lI[r];
#pragma unroll
        for (int dt = 0; dt < 4; ++dt) {
            int d = dt * 16 + l16;
            float o = oacc[dt][r] * inv;
#pragma unroll
            for (int j = 0; j < NBUCKET; ++j)
                o += bk5[r][j] * inv * ldsRV[j * DHEAD + d];
            Ctx[((size_t)(bI * S_LEN + qg)) * DMODEL + hI * DHEAD + d] = (__bf16)o;
        }
    }
}

// ---------------------------------------------------------------------------
// Output projection: Ctx[4096x1024] @ Wo[1024x1024] + bo -> fp32 out
// ---------------------------------------------------------------------------
__global__ __launch_bounds__(128)
void gemm_out_kernel(const __bf16* __restrict__ A, const __bf16* __restrict__ W,
                     const float* __restrict__ bias, float* __restrict__ Out) {
    __shared__ __bf16 ldsA[2][64 * 32];
    __shared__ __bf16 ldsBt[2][64 * 32];

    const int tid  = threadIdx.x;
    const int lane = tid & 31, wave = tid >> 5;
    const int wm = wave >> 1, wn = wave & 1;
    const int hlf = lane >> 4, l16 = lane & 15;
    const int m0 = blockIdx.x * 64, n0 = blockIdx.y * 64;

    v8f acc[2][2] = {};

    auto stage = [&](int buf, int k0) {
        {
            int row = tid >> 1, h2 = tid & 1;
            const __bf16* gA = A + (size_t)(m0 + row) * DMODEL + k0 + h2 * 16;
            __bf16* lA = &ldsA[buf][row * 32 + h2 * 16];
            async_ld_b128(lA, gA);
            async_ld_b128(lA + 8, gA + 8);
        }
        {
            int kr = tid >> 2, qt = tid & 3;
            union { uint4 q[2]; __bf16 h[16]; } bufr;
            const uint4* src = (const uint4*)(W + (size_t)(k0 + kr) * DMODEL + n0 + qt * 16);
            bufr.q[0] = src[0]; bufr.q[1] = src[1];
#pragma unroll
            for (int e = 0; e < 16; ++e)
                ldsBt[buf][(qt * 16 + e) * 32 + kr] = bufr.h[e];
        }
    };

    stage(0, 0);
    wait_async();
    __syncthreads();

    for (int k0 = 0; k0 < DMODEL; k0 += 32) {
        const int cur = (k0 >> 5) & 1;
        if (k0 + 32 < DMODEL) stage(1 - cur, k0 + 32);

        AF a[2], b[2];
#pragma unroll
        for (int sm = 0; sm < 2; ++sm) {
            int mr = wm * 32 + sm * 16 + l16;
#pragma unroll
            for (int j = 0; j < 8; ++j)
                a[sm].u[j] = *(const unsigned int*)&ldsA[cur][mr * 32 + kbA(j, hlf)];
        }
#pragma unroll
        for (int sn = 0; sn < 2; ++sn) {
            int nc = wn * 32 + sn * 16 + l16;
#pragma unroll
            for (int j = 0; j < 8; ++j)
                b[sn].u[j] = *(const unsigned int*)&ldsBt[cur][nc * 32 + kbB(j, hlf)];
        }
#pragma unroll
        for (int sm = 0; sm < 2; ++sm)
#pragma unroll
            for (int sn = 0; sn < 2; ++sn)
                acc[sm][sn] = __builtin_amdgcn_wmma_f32_16x16x32_bf16(
                    false, a[sm].v, false, b[sn].v, (short)0, acc[sm][sn], false, false);

        wait_async();
        __syncthreads();
    }

#pragma unroll
    for (int sm = 0; sm < 2; ++sm)
#pragma unroll
        for (int sn = 0; sn < 2; ++sn)
#pragma unroll
            for (int r = 0; r < 8; ++r) {
                int m = m0 + wm * 32 + sm * 16 + r + 8 * hlf;
                int n = n0 + wn * 32 + sn * 16 + l16;
                Out[(size_t)m * DMODEL + n] = acc[sm][sn][r] + bias[n];
            }
}

// ---------------------------------------------------------------------------
extern "C" void kernel_launch(void* const* d_in, const int* in_sizes, int n_in,
                              void* d_out, int out_size, void* d_ws, size_t ws_size,
                              hipStream_t stream) {
    const float* x     = (const float*)d_in[0];
    const float* Wq    = (const float*)d_in[1];
    const float* bq    = (const float*)d_in[2];
    const float* Wk    = (const float*)d_in[3];
    const float* bk    = (const float*)d_in[4];
    const float* Wv    = (const float*)d_in[5];
    const float* bv    = (const float*)d_in[6];
    const float* Wo    = (const float*)d_in[7];
    const float* bo    = (const float*)d_in[8];
    const float* rel_k = (const float*)d_in[9];
    const float* rel_v = (const float*)d_in[10];
    (void)in_sizes; (void)n_in; (void)out_size; (void)ws_size;

    // workspace partition (48 MB total)
    char* ws = (char*)d_ws;
    __bf16* xb  = (__bf16*)ws;  ws += (size_t)MROWS * DMODEL * 2;
    __bf16* wqb = (__bf16*)ws;  ws += (size_t)DMODEL * DMODEL * 2;
    __bf16* wkb = (__bf16*)ws;  ws += (size_t)DMODEL * DMODEL * 2;
    __bf16* wvb = (__bf16*)ws;  ws += (size_t)DMODEL * DMODEL * 2;
    __bf16* wob = (__bf16*)ws;  ws += (size_t)DMODEL * DMODEL * 2;
    __bf16* qb  = (__bf16*)ws;  ws += (size_t)MROWS * DMODEL * 2;
    __bf16* kb  = (__bf16*)ws;  ws += (size_t)MROWS * DMODEL * 2;
    __bf16* vb  = (__bf16*)ws;  ws += (size_t)MROWS * DMODEL * 2;
    __bf16* ctx = (__bf16*)ws;  ws += (size_t)MROWS * DMODEL * 2;

    const int nX4 = MROWS * DMODEL / 4;      // 1048576 groups of 4
    const int nW4 = DMODEL * DMODEL / 4;     // 262144 groups of 4
    cvt_bf16_kernel<<<(nX4 + 255) / 256, 256, 0, stream>>>((const float4*)x,  (uint2*)xb,  nX4);
    cvt_bf16_kernel<<<(nW4 + 255) / 256, 256, 0, stream>>>((const float4*)Wq, (uint2*)wqb, nW4);
    cvt_bf16_kernel<<<(nW4 + 255) / 256, 256, 0, stream>>>((const float4*)Wk, (uint2*)wkb, nW4);
    cvt_bf16_kernel<<<(nW4 + 255) / 256, 256, 0, stream>>>((const float4*)Wv, (uint2*)wvb, nW4);
    cvt_bf16_kernel<<<(nW4 + 255) / 256, 256, 0, stream>>>((const float4*)Wo, (uint2*)wob, nW4);

    gemm_qkv_kernel<<<dim3(MROWS / 64, DMODEL / 64, 3), 128, 0, stream>>>(
        xb, wqb, wkb, wvb, bq, bk, bv, qb, kb, vb);

    attn_kernel<<<dim3(S_LEN / 64, BATCH * NHEAD), 128, 0, stream>>>(
        qb, kb, vb, rel_k, rel_v, ctx);

    gemm_out_kernel<<<dim3(MROWS / 64, DMODEL / 64), 128, 0, stream>>>(
        ctx, wob, bo, (float*)d_out);
}
